// PEARLPositionalEncoder_77532749627610
// MI455X (gfx1250) — matrix operand order repaired and checked by reading, not compiled, not run on hardware
//
#include <hip/hip_runtime.h>

typedef __attribute__((ext_vector_type(16))) _Float16 v16h;
typedef __attribute__((ext_vector_type(8)))  _Float16 h8;
typedef __attribute__((ext_vector_type(8)))  float    v8f;

#define NB    128        // graphs
#define NN    128        // nodes per graph
#define NM    16         // samples
#define NK    8          // pearl_k
#define NH    16         // mlp hidden
#define ND    64         // pe dims
#define NE    262144     // edges
#define NSUM  (NB*NN)    // 16384 nodes total

static __device__ __forceinline__ v8f wmma_f16(v16h a, v16h b, v8f c) {
    return __builtin_amdgcn_wmma_f32_16x16x32_f16(false, a, false, b, (short)0, c, false, false);
}
static __device__ __forceinline__ v16h cat8(h8 lo, h8 hi) {
    return __builtin_shufflevector(lo, hi, 0,1,2,3,4,5,6,7,8,9,10,11,12,13,14,15);
}

// ---------------------------------------------------------------------------
// Prep: convert GIN weights f32 -> f16 once, stored TRANSPOSED ([col][k]) so
// WMMA B-fragments are K-contiguous per lane (vector LDS loads downstream).
// Layout in wts: [W1a^T 64x16 | W1b^T 64x64 | W2a^T 64x64 | W2b^T 64x64]
// ---------------------------------------------------------------------------
__global__ __launch_bounds__(256) void prep_weights_kernel(
    const float* __restrict__ W1a, const float* __restrict__ W1b,
    const float* __restrict__ W2a, const float* __restrict__ W2b,
    _Float16* __restrict__ wts)
{
    int i = blockIdx.x * 256 + threadIdx.x;           // 13312 total
    if (i < 1024) {                                   // W1a^T : [c][k] k<16
        int c = i >> 4, k = i & 15;
        wts[i] = (_Float16)W1a[k * ND + c];
    } else if (i < 5120) {                            // W1b^T : [c][k] k<64
        int j = i - 1024, c = j >> 6, k = j & 63;
        wts[i] = (_Float16)W1b[k * ND + c];
    } else if (i < 9216) {                            // W2a^T
        int j = i - 5120, c = j >> 6, k = j & 63;
        wts[i] = (_Float16)W2a[k * ND + c];
    } else if (i < 13312) {                           // W2b^T
        int j = i - 9216, c = j >> 6, k = j & 63;
        wts[i] = (_Float16)W2b[k * ND + c];
    }
}

// ---------------------------------------------------------------------------
// Stage 1: polynomial filter (7x [128x128]@[128x16] per graph, WMMA) fused with
// k->H MLP (WMMA, K padded 8->32), per-graph BatchNorm, ReLU.
// One workgroup (8 waves) per graph. mlp_b omitted (cancels in BN mean-subtract).
// featT[buf][m][n]   : transposed current feature  -> B-frag = 2x ds_load_b128
// featA[n][m][k(8)]  : K-contiguous copy           -> 1b A-frag = 1x ds_load_b128
// ---------------------------------------------------------------------------
__global__ __launch_bounds__(256) void poly_mlp_bn_kernel(
    const float* __restrict__ lap, const float* __restrict__ W0,
    const float* __restrict__ mlpW, const float* __restrict__ bng,
    const float* __restrict__ bnb, _Float16* __restrict__ hbuf0)
{
    __shared__ _Float16 featT[2][NM][NN];   // 8 KB double buffer
    __shared__ _Float16 featA[NN][NM][NK];  // 32 KB
    __shared__ float    stats[2 * NH];

    const int b     = blockIdx.x;
    const int tid   = threadIdx.x;
    const int wave  = tid >> 5;
    const int lane  = tid & 31;
    const int row   = lane & 15;       // A row / B,D col
    const int khalf = lane >> 4;
    const int col   = lane & 15;
    const int kb16  = khalf * 16;

    const float* lapg = lap + (size_t)b * NN * NN;
    const float* w0g  = W0  + (size_t)b * NN * NM;

    for (int i = tid; i < NN * NM; i += 256) {
        int n = i >> 4, m = i & 15;
        _Float16 v = (_Float16)w0g[i];
        featT[0][m][n] = v;
        featA[n][m][0] = v;
    }
    if (tid < 2 * NH) stats[tid] = 0.f;
    __syncthreads();

    // --- polynomial filter: feat[k] = lap @ feat[k-1] -----------------------
    for (int k = 1; k < NK; ++k) {
        const int cur = (k - 1) & 1, nxt = k & 1;
        v8f acc = {};
        #pragma unroll
        for (int kk = 0; kk < 4; ++kk) {
            // A: lap rows, two contiguous 8-float runs -> 4x global_load_b128
            const float4* lp =
                (const float4*)(lapg + (wave * 16 + row) * NN + kk * 32 + (khalf << 3));
            float4 f0 = lp[0], f1 = lp[1], f2 = lp[4], f3 = lp[5];
            v16h A;
            A[0]=(_Float16)f0.x; A[1]=(_Float16)f0.y; A[2]=(_Float16)f0.z; A[3]=(_Float16)f0.w;
            A[4]=(_Float16)f1.x; A[5]=(_Float16)f1.y; A[6]=(_Float16)f1.z; A[7]=(_Float16)f1.w;
            A[8]=(_Float16)f2.x; A[9]=(_Float16)f2.y; A[10]=(_Float16)f2.z; A[11]=(_Float16)f2.w;
            A[12]=(_Float16)f3.x; A[13]=(_Float16)f3.y; A[14]=(_Float16)f3.z; A[15]=(_Float16)f3.w;
            // B: featT K-contiguous -> 2x ds_load_b128
            h8 b0 = *(const h8*)&featT[cur][col][kk * 32 + khalf * 8];
            h8 b1 = *(const h8*)&featT[cur][col][kk * 32 + khalf * 8 + 16];
            acc = wmma_f16(A, cat8(b0, b1), acc);
        }
        // D -> featT (vector) + featA (scatter)
        h8 o;
        #pragma unroll
        for (int r = 0; r < 8; ++r) o[r] = (_Float16)acc[r];
        *(h8*)&featT[nxt][col][wave * 16 + khalf * 8] = o;
        #pragma unroll
        for (int r = 0; r < 8; ++r)
            featA[wave * 16 + khalf * 8 + r][col][k] = o[r];
        __syncthreads();
    }

    // --- k->H MLP per node: h[n] = feat[n](16m x 8k, pad K->32) @ mlpW ------
    v16h Bm;                                        // built once, reused 16x
    #pragma unroll
    for (int e = 0; e < 16; ++e) {
        int kb = kb16 + e;
        Bm[e] = (kb < NK) ? (_Float16)mlpW[kb * NH + col] : (_Float16)0.f;
    }

    float acc2[16][8];
    #pragma unroll
    for (int t = 0; t < 16; ++t) {
        int n = wave * 16 + t;
        h8 alo = {};
        if (khalf == 0) alo = *(const h8*)&featA[n][row][0];  // k = 0..7
        v8f c = {};
        c = wmma_f16(cat8(alo, (h8){}), Bm, c);
        #pragma unroll
        for (int r = 0; r < 8; ++r) acc2[t][r] = c[r];
    }

    // --- BN stats over (n,m) = 2048 values per channel ----------------------
    float s = 0.f, s2 = 0.f;
    #pragma unroll
    for (int t = 0; t < 16; ++t)
        #pragma unroll
        for (int r = 0; r < 8; ++r) { float x = acc2[t][r]; s += x; s2 += x * x; }
    atomicAdd(&stats[col], s);
    atomicAdd(&stats[NH + col], s2);
    __syncthreads();

    const float mu  = stats[col] * (1.f / 2048.f);
    const float var = stats[NH + col] * (1.f / 2048.f) - mu * mu;
    const float inv = rsqrtf(var + 1e-5f) * bng[col];
    const float bet = bnb[col];

    #pragma unroll
    for (int t = 0; t < 16; ++t) {
        int n = wave * 16 + t;
        #pragma unroll
        for (int r = 0; r < 8; ++r) {
            int m = r + khalf * 8;
            float y = fmaxf((acc2[t][r] - mu) * inv + bet, 0.f);
            hbuf0[(((size_t)b * NN + n) * NM + m) * NH + col] = (_Float16)y;
        }
    }
}

// ---------------------------------------------------------------------------
// Stage 2: edge aggregation  agg[dst] += h[src]  with packed-f16 atomics.
// One thread per 8 f16 elements of one edge: 1x b128 load + 4x pk_add_f16.
// ---------------------------------------------------------------------------
template <int ELEMS>
__global__ __launch_bounds__(256) void gin_agg_kernel(
    const _Float16* __restrict__ h, const int* __restrict__ ei,
    _Float16* __restrict__ agg)
{
    constexpr int OCTS = ELEMS / 8;
    size_t gid = (size_t)blockIdx.x * 256 + threadIdx.x;
    int e = (int)(gid / OCTS);
    int q = (int)(gid % OCTS);
    int src = ei[e];
    int dst = ei[NE + e];
    const uint4 d = *(const uint4*)(h + (size_t)src * ELEMS + q * 8);
    uint64_t a64 = (uint64_t)(uintptr_t)(agg + (size_t)dst * ELEMS + q * 8);
    asm volatile(
        "global_atomic_pk_add_f16 %0, %1, off\n\t"
        "global_atomic_pk_add_f16 %0, %2, off offset:4\n\t"
        "global_atomic_pk_add_f16 %0, %3, off offset:8\n\t"
        "global_atomic_pk_add_f16 %0, %4, off offset:12"
        :: "v"(a64), "v"(d.x), "v"(d.y), "v"(d.z), "v"(d.w) : "memory");
}

// ---------------------------------------------------------------------------
// Stage 3: GIN node MLP. One wave per node (node's M=16 samples = one WMMA
// 16-row tile). Pre-transposed f16 weights pulled to LDS with CDNA5 async
// global->LDS copies. FINAL variant fuses the sample-dim reduction and writes
// f32 output directly.
// ---------------------------------------------------------------------------
template <int DIN, bool FINAL>
__global__ __launch_bounds__(256) void gin_mlp_kernel(
    const _Float16* __restrict__ hin, const _Float16* __restrict__ agg,
    const float* __restrict__ epsp,
    const _Float16* __restrict__ wtsT,   // [WaT (ND x DIN) | WbT (ND x ND)]
    const float* __restrict__ ba, const float* __restrict__ bb,
    _Float16* __restrict__ hout, float* __restrict__ outF)
{
    constexpr int KT1    = (DIN + 31) / 32;
    constexpr int WHALF  = ND * DIN + ND * ND;        // f16 elements of weights
    constexpr int CHUNKS = WHALF / 8;                 // 16B chunks
    __shared__ _Float16 WL[WHALF];                    // WaT | WbT
    __shared__ float    baL[ND], bbL[ND];
    __shared__ _Float16 t1L[8][NM * ND];              // per-wave scratch

    const int tid   = threadIdx.x;
    const int wave  = tid >> 5;
    const int lane  = tid & 31;
    const int row   = lane & 15;
    const int khalf = lane >> 4;
    const int col   = lane & 15;
    const int kb16  = khalf * 16;

    // --- async DMA of pre-converted weights into LDS (gfx1250 path) --------
    {
        const uint32_t ldsW  = (uint32_t)(uintptr_t)&WL[0];
        const uint64_t gbase = (uint64_t)(uintptr_t)wtsT;
        for (int i = tid; i < CHUNKS; i += 256) {
            uint32_t la = ldsW + (uint32_t)(i * 16);
            uint64_t ga = gbase + (uint64_t)(i * 16);
            asm volatile("global_load_async_to_lds_b128 %0, %1, off"
                         :: "v"(la), "v"(ga) : "memory");
        }
    }
    if (tid < ND) { baL[tid] = ba[tid]; bbL[tid] = bb[tid]; }
    const _Float16 eps1p = (_Float16)(1.f + epsp[0]);
    asm volatile("s_wait_asynccnt 0x0" ::: "memory");
    __syncthreads();

    const _Float16* WaT = WL;                // [c][k], row length DIN
    const _Float16* WbT = WL + ND * DIN;     // [c][k], row length ND

    const size_t nd = (size_t)blockIdx.x * 8 + wave;
    const _Float16* hp = hin + nd * (NM * DIN);
    const _Float16* ap = agg + nd * (NM * DIN);

    // --- z = (1+eps)*h + agg fragments (vector global loads, pk f16 fma) ----
    v16h Az[KT1];
    #pragma unroll
    for (int kt = 0; kt < KT1; ++kt) {
        int base = kt * 32 + khalf * 8;
        h8 lo = {}, hi = {};
        if (base < DIN)
            lo = *(const h8*)(hp + row * DIN + base) * eps1p +
                 *(const h8*)(ap + row * DIN + base);
        if (base + 16 < DIN)
            hi = *(const h8*)(hp + row * DIN + base + 16) * eps1p +
                 *(const h8*)(ap + row * DIN + base + 16);
        Az[kt] = cat8(lo, hi);
    }

    // --- t1 = relu(z @ Wa + ba) --------------------------------------------
    #pragma unroll
    for (int ct = 0; ct < 4; ++ct) {
        v8f acc = {};
        #pragma unroll
        for (int kt = 0; kt < KT1; ++kt) {
            h8 b0 = {}, b1 = {};
            if (kt * 32 + khalf * 8 < DIN)
                b0 = *(const h8*)&WaT[(ct * 16 + col) * DIN + kt * 32 + khalf * 8];
            if (kt * 32 + khalf * 8 + 16 < DIN)
                b1 = *(const h8*)&WaT[(ct * 16 + col) * DIN + kt * 32 + khalf * 8 + 16];
            acc = wmma_f16(Az[kt], cat8(b0, b1), acc);
        }
        #pragma unroll
        for (int r = 0; r < 8; ++r) {
            int m = r + khalf * 8;
            float y = fmaxf(acc[r] + baL[ct * 16 + col], 0.f);
            t1L[wave][m * ND + ct * 16 + col] = (_Float16)y;
        }
    }
    __syncthreads();

    // --- A2 fragments from t1 (K-contiguous -> vector LDS loads) ------------
    v16h A2[2];
    #pragma unroll
    for (int kt = 0; kt < 2; ++kt) {
        h8 lo = *(const h8*)&t1L[wave][row * ND + kt * 32 + khalf * 8];
        h8 hi = *(const h8*)&t1L[wave][row * ND + kt * 32 + khalf * 8 + 16];
        A2[kt] = cat8(lo, hi);
    }

    // --- out = relu(t1 @ Wb + bb) ------------------------------------------
    #pragma unroll
    for (int ct = 0; ct < 4; ++ct) {
        v8f acc = {};
        #pragma unroll
        for (int kt = 0; kt < 2; ++kt) {
            h8 b0 = *(const h8*)&WbT[(ct * 16 + col) * ND + kt * 32 + khalf * 8];
            h8 b1 = *(const h8*)&WbT[(ct * 16 + col) * ND + kt * 32 + khalf * 8 + 16];
            acc = wmma_f16(A2[kt], cat8(b0, b1), acc);
        }
        if constexpr (FINAL) {
            // fused RandomSampleAggregator: sum over M inside the tile
            float s = 0.f;
            #pragma unroll
            for (int r = 0; r < 8; ++r)
                s += fmaxf(acc[r] + bbL[ct * 16 + col], 0.f);
            float tot = s + __shfl_xor(s, 16);
            if (khalf == 0)
                outF[nd * ND + ct * 16 + col] = tot;
        } else {
            #pragma unroll
            for (int r = 0; r < 8; ++r) {
                int m = r + khalf * 8;
                float y = fmaxf(acc[r] + bbL[ct * 16 + col], 0.f);
                t1L[wave][m * ND + ct * 16 + col] = (_Float16)y;  // stage for b128 store
            }
        }
    }
    if constexpr (!FINAL) {
        // coalesced vector store of the node tile
        #pragma unroll
        for (int it = 0; it < 4; ++it) {
            int off = it * 256 + lane * 8;
            h8 v = *(const h8*)&t1L[wave][off];
            *(h8*)(hout + nd * (NM * ND) + off) = v;
        }
    }
}

// ---------------------------------------------------------------------------
extern "C" void kernel_launch(void* const* d_in, const int* in_sizes, int n_in,
                              void* d_out, int out_size, void* d_ws, size_t ws_size,
                              hipStream_t stream) {
    const float* lap  = (const float*)d_in[0];
    const float* W0   = (const float*)d_in[1];
    const float* mlpW = (const float*)d_in[2];
    // d_in[3] = mlp_b : unused (cancels inside BatchNorm mean subtraction)
    const float* bng  = (const float*)d_in[4];
    const float* bnb  = (const float*)d_in[5];
    const float* eps1 = (const float*)d_in[6];
    const float* W1a  = (const float*)d_in[7];
    const float* b1a  = (const float*)d_in[8];
    const float* W1b  = (const float*)d_in[9];
    const float* b1b  = (const float*)d_in[10];
    const float* eps2 = (const float*)d_in[11];
    const float* W2a  = (const float*)d_in[12];
    const float* b2a  = (const float*)d_in[13];
    const float* W2b  = (const float*)d_in[14];
    const float* b2b  = (const float*)d_in[15];
    const int*   ei   = (const int*)d_in[16];

    // workspace layout (~80 MB)
    char* ws = (char*)d_ws;
    _Float16* h0   = (_Float16*)(ws);                       //  8 MB [NSUM,16,16]
    _Float16* agg1 = (_Float16*)(ws + ((size_t)8  << 20));  //  8 MB [NSUM,16,16]
    _Float16* h1   = (_Float16*)(ws + ((size_t)16 << 20));  // 32 MB [NSUM,16,64]
    _Float16* agg2 = (_Float16*)(ws + ((size_t)48 << 20));  // 32 MB [NSUM,16,64]
    _Float16* wts  = (_Float16*)(ws + ((size_t)80 << 20));  // 26 KB transposed f16 weights

    hipMemsetAsync(agg1, 0, (size_t)8  << 20, stream);
    hipMemsetAsync(agg2, 0, (size_t)32 << 20, stream);

    prep_weights_kernel<<<52, 256, 0, stream>>>(W1a, W1b, W2a, W2b, wts);
    poly_mlp_bn_kernel<<<NB, 256, 0, stream>>>(lap, W0, mlpW, bng, bnb, h0);

    gin_agg_kernel<NM * NH><<<NE * (NM * NH / 8) / 256, 256, 0, stream>>>(h0, ei, agg1);
    gin_mlp_kernel<NH, false><<<NSUM / 8, 256, 0, stream>>>(
        h0, agg1, eps1, wts, b1a, b1b, h1, nullptr);

    gin_agg_kernel<NM * ND><<<NE * (NM * ND / 8) / 256, 256, 0, stream>>>(h1, ei, agg2);
    gin_mlp_kernel<ND, true><<<NSUM / 8, 256, 0, stream>>>(
        h1, agg2, eps2, wts + 5120, b2a, b2b, nullptr, (float*)d_out);
}
